// NN_45372034515615
// MI455X (gfx1250) — matrix operand order repaired and checked by reading, not compiled, not run on hardware
//
#include <hip/hip_runtime.h>
#include <hip/hip_bf16.h>

// ---------------------------------------------------------------------------
// Problem constants (from reference)
// ---------------------------------------------------------------------------
#define HD    128     // hidden
#define SS    29      // sequence
#define NBAT  8192    // batch
#define NHEAD 2       // hard-attention heads
#define ADIM  256     // per-head attention dim (512 / 2)
#define NCODE 100     // SimVQ codebook size
#define NCPAD 112     // 100 padded up to 7*16 tiles
#define NROW  (NBAT*NHEAD)  // 16384 quantized rows
#define GB    16      // batch rows per attention block

typedef __attribute__((ext_vector_type(16))) _Float16 v16h;
typedef __attribute__((ext_vector_type(8)))  float    v8f;

// ---------------------------------------------------------------------------
// WMMA fragment K-index maps (CDNA5 16-bit 16x32 A / 32x16 B layouts)
// A: lane l -> M = l%16 ; half j -> K = 8*(l/16) + j + (j<8 ? 0 : 8)
// B: lane l -> N = l%16 ; half j -> K = 8*(l/16) + j + (j<8 ? 0 : 8)
// ---------------------------------------------------------------------------
__device__ __forceinline__ int a_kidx(int g, int j) { return 8*g + j + ((j < 8) ? 0 : 8); }
__device__ __forceinline__ int b_kidx(int g, int j) { return 8*g + j + ((j < 8) ? 0 : 8); }

__device__ __forceinline__ float wave_sum(float v) {
#pragma unroll
  for (int off = 16; off > 0; off >>= 1) v += __shfl_xor(v, off, 32);
  return v;
}

// ---------------------------------------------------------------------------
// K1: Mht[n][k][h1] = (Wq_n Wk_n^T)[h1,k] as f16, transposed for coalesced
// B-fragment loads in the attention kernel. One wave per 16x16 tile.
// ---------------------------------------------------------------------------
__global__ void k_prep_M(const float* __restrict__ Wq, const float* __restrict__ Wk,
                         _Float16* __restrict__ Mht) {
  const int blk  = blockIdx.x;
  const int n    = blk >> 6;          // head
  const int t    = blk & 63;
  const int tm   = (t >> 3) * 16;     // h1 tile
  const int tn   = (t & 7) * 16;      // h2 (=k) tile
  const int lane = threadIdx.x;
  const int g = lane >> 4, lr = lane & 15;
  const int base = n * ADIM;

  v8f acc = {};
  for (int k0 = 0; k0 < ADIM; k0 += 32) {
    v16h a, b;
    const int row = tm + lr;          // h1
    const int col = tn + lr;          // h2
#pragma unroll
    for (int j = 0; j < 16; ++j)
      a[j] = (_Float16)Wq[row * 512 + base + k0 + a_kidx(g, j)];
#pragma unroll
    for (int j = 0; j < 16; ++j)
      b[j] = (_Float16)Wk[col * 512 + base + k0 + b_kidx(g, j)];
    acc = __builtin_amdgcn_wmma_f32_16x16x32_f16(false, a, false, b, (short)0, acc,
                                                 false, false);
  }
  // transposed store: Mht[n][k = tn+lr][h1 = tm+r+8g]
#pragma unroll
  for (int r = 0; r < 8; ++r)
    Mht[n * HD * HD + (tn + lr) * HD + (tm + r + 8 * g)] = (_Float16)acc[r];
}

// ---------------------------------------------------------------------------
// K2: implicit codebook cb = code_embed @ Wc  (100x128, padded to 112 rows)
// ---------------------------------------------------------------------------
__global__ void k_prep_cb(const float* __restrict__ ce, const float* __restrict__ Wc,
                          float* __restrict__ cb, _Float16* __restrict__ cbh) {
  const int blk  = blockIdx.x;        // 7 row tiles * 8 col tiles = 56
  const int tm   = (blk >> 3) * 16;   // code tile
  const int tn   = (blk & 7) * 16;    // h tile
  const int lane = threadIdx.x;
  const int g = lane >> 4, lr = lane & 15;

  v8f acc = {};
  for (int k0 = 0; k0 < HD; k0 += 32) {
    v16h a, b;
    const int crow = tm + lr;
    const int hcol = tn + lr;
#pragma unroll
    for (int j = 0; j < 16; ++j) {
      const int kk = k0 + a_kidx(g, j);
      a[j] = (crow < NCODE) ? (_Float16)ce[crow * HD + kk] : (_Float16)0.0f;
    }
#pragma unroll
    for (int j = 0; j < 16; ++j) {
      const int kk = k0 + b_kidx(g, j);
      b[j] = (_Float16)Wc[kk * HD + hcol];   // B[k][h] = Wc[k,h]
    }
    acc = __builtin_amdgcn_wmma_f32_16x16x32_f16(false, a, false, b, (short)0, acc,
                                                 false, false);
  }
#pragma unroll
  for (int r = 0; r < 8; ++r) {
    const int crow = tm + r + 8 * g;
    const int hcol = tn + lr;
    cb [crow * HD + hcol] = acc[r];
    cbh[crow * HD + hcol] = (_Float16)acc[r];
  }
}

// ---------------------------------------------------------------------------
// K3: codebook squared norms, positional-encoding table, zero loss slot.
// ---------------------------------------------------------------------------
__global__ void k_prep_aux(const float* __restrict__ cb, float* __restrict__ cbn2,
                           float* __restrict__ pet, float* __restrict__ loss) {
  const int t = threadIdx.x;          // 128 threads
  // pe table column t
  const float i2   = (float)(t & ~1);
  const float freq = __expf(-logf(10000.0f) * i2 / (float)HD);
  const bool  iscos = (t & 1);
  for (int s = 0; s < SS; ++s) {
    const float ph = (float)s * freq;
    pet[s * HD + t] = iscos ? __cosf(ph) : __sinf(ph);
  }
  if (t < NCPAD) {
    if (t < NCODE) {
      float s = 0.f;
      for (int h = 0; h < HD; ++h) { float v = cb[t * HD + h]; s += v * v; }
      cbn2[t] = s;
    } else {
      cbn2[t] = 1e30f;
    }
  }
  if (t == 0) *loss = 0.f;
}

// ---------------------------------------------------------------------------
// K4: fused attention pass, GB=16 batch rows per 256-thread block.
//   stage1: stream x (+pe table) -> f16 xp in LDS, f32 xsum
//   stage2: V[16,2*128] = Xsum @ Mht via v_wmma (16 tiles x 4 k-steps)
//   stage3: logits[b,n,s] = xp_s . v_n /(sqrtH*S) + gumbel ; argmax
//   stage4: gather raw x rows -> xa
// ---------------------------------------------------------------------------
__global__ void __launch_bounds__(256) k_attend(
    const float* __restrict__ x, const float* __restrict__ gu,
    const _Float16* __restrict__ Mht, const float* __restrict__ pet,
    float* __restrict__ xa) {
  __shared__ _Float16 xph[GB][SS][HD];        // 118.8 KB
  __shared__ float    xsum[GB][HD + 4];       // padded vs bank conflicts
  __shared__ float    vv[GB][NHEAD * HD];
  __shared__ float    sc[GB][NHEAD][32];
  __shared__ int      sidx[GB][NHEAD];

  const int tid = threadIdx.x;
  const int b0  = blockIdx.x * GB;

  // ---- stage 1: stream + column sums (each 32-thread group owns 2 b's) ----
  {
    const int grp = tid >> 5;                 // 0..7
    const int h0  = (tid & 31) * 4;           // 4 columns per thread
#pragma unroll
    for (int bi = 0; bi < 2; ++bi) {
      const int bl = grp * 2 + bi;
      const float* xb = x + (size_t)(b0 + bl) * SS * HD;
      float s0 = 0.f, s1 = 0.f, s2 = 0.f, s3 = 0.f;
      for (int s = 0; s < SS; ++s) {
        const float4 xv = *(const float4*)(xb + s * HD + h0);
        const float4 pv = *(const float4*)(pet + s * HD + h0);
        const float a0 = xv.x + pv.x, a1 = xv.y + pv.y;
        const float a2 = xv.z + pv.z, a3 = xv.w + pv.w;
        xph[bl][s][h0 + 0] = (_Float16)a0;
        xph[bl][s][h0 + 1] = (_Float16)a1;
        xph[bl][s][h0 + 2] = (_Float16)a2;
        xph[bl][s][h0 + 3] = (_Float16)a3;
        s0 += a0; s1 += a1; s2 += a2; s3 += a3;
      }
      xsum[bl][h0 + 0] = s0; xsum[bl][h0 + 1] = s1;
      xsum[bl][h0 + 2] = s2; xsum[bl][h0 + 3] = s3;
    }
    if (b0 + GB < NBAT)                       // warm GL2 for the next block
      __builtin_prefetch(x + (size_t)(b0 + GB) * SS * HD + tid * 16, 0, 3);
  }
  __syncthreads();

  // ---- stage 2: V = Xsum[16,128] @ Mht (WMMA, two 16x16 tiles per wave) ----
  {
    const int wave = tid >> 5;                // 0..7
    const int lane = tid & 31;
    const int g = lane >> 4, lr = lane & 15;
#pragma unroll
    for (int t = wave; t < 16; t += 8) {      // 2 tiles per wave
      const int n  = t >> 3;                  // head
      const int tn = (t & 7) * 16;            // h1 tile
      v8f acc = {};
      for (int k0 = 0; k0 < HD; k0 += 32) {
        v16h a, b;
#pragma unroll
        for (int j = 0; j < 16; ++j)          // A[b_row=lr][k]
          a[j] = (_Float16)xsum[lr][k0 + a_kidx(g, j)];
#pragma unroll
        for (int j = 0; j < 16; ++j)          // B[k][h1] = Mht[n][k][h1], coalesced
          b[j] = Mht[n * HD * HD + (k0 + b_kidx(g, j)) * HD + tn + lr];
        acc = __builtin_amdgcn_wmma_f32_16x16x32_f16(false, a, false, b, (short)0, acc,
                                                     false, false);
      }
#pragma unroll
      for (int r = 0; r < 8; ++r)
        vv[r + 8 * g][n * HD + tn + lr] = acc[r];
    }
  }
  __syncthreads();

  // ---- stage 3: logits + gumbel; 928 dot-jobs over 256 threads ----
  const float inv = 1.0f / (sqrtf((float)HD) * (float)SS);
  for (int j = tid; j < GB * NHEAD * SS; j += 256) {
    const int bl  = j / (NHEAD * SS);
    const int rem = j % (NHEAD * SS);
    const int n = rem / SS, s = rem % SS;
    float acc = 0.f;
    const int h0s = (tid * 4) & (HD - 1);     // stagger LDS banks
    for (int i = 0; i < HD; ++i) {
      const int h = (h0s + i) & (HD - 1);
      acc += (float)xph[bl][s][h] * vv[bl][n * HD + h];
    }
    const float u  = gu[((size_t)(b0 + bl) * NHEAD + n) * SS + s];
    sc[bl][n][s] = acc * inv + (-logf(-logf(u)));   // /TAU monotone -> skip
  }
  __syncthreads();

  if (tid < GB * NHEAD) {                     // first-max argmax (matches jnp)
    const int bl = tid >> 1, n = tid & 1;
    int best = 0; float bv = sc[bl][n][0];
    for (int s = 1; s < SS; ++s) {
      const float v = sc[bl][n][s];
      if (v > bv) { bv = v; best = s; }
    }
    sidx[bl][n] = best;
  }
  __syncthreads();

  // ---- stage 4: gather raw x rows (exact fp32) -> xa ----
  {
    const int rr = tid >> 3;                  // 0..31 output row
    const int bl = rr >> 1, n = rr & 1;
    const int s  = sidx[bl][n];
    const float* src = x  + ((size_t)(b0 + bl) * SS + s) * HD;
    float*       dst = xa + ((size_t)(b0 + bl) * NHEAD + n) * HD;
#pragma unroll
    for (int j = 0; j < 4; ++j) {
      const int c = ((tid & 7) + 8 * j) * 4;
      *(float4*)(dst + c) = *(const float4*)(src + c);
    }
  }
}

// ---------------------------------------------------------------------------
// K5: SimVQ nearest code (WMMA distance GEMM, 16 rows x 112 codes per wave)
//     + rotation-trick output + commit loss. One wave per 16 rows.
// ---------------------------------------------------------------------------
__global__ void k_vq(const float* __restrict__ xa, const float* __restrict__ cb,
                     const _Float16* __restrict__ cbh, const float* __restrict__ cbn2,
                     float* __restrict__ q, float* __restrict__ idxf,
                     float* __restrict__ loss) {
  __shared__ float dist[16 * NCPAD];
  __shared__ int   ridx[16];

  const int lane = threadIdx.x;
  const int g = lane >> 4, lr = lane & 15;
  const int row0 = blockIdx.x * 16;

  // dot(flat_row, cb_c) for 16 rows x 112 codes via v_wmma_f32_16x16x32_f16
  for (int tn = 0; tn < 7; ++tn) {
    v8f acc = {};
    for (int k0 = 0; k0 < HD; k0 += 32) {
      v16h a, b;
      const int row = row0 + lr;
      const int c   = tn * 16 + lr;
#pragma unroll
      for (int j = 0; j < 16; ++j)
        a[j] = (_Float16)xa[(size_t)row * HD + k0 + a_kidx(g, j)];
#pragma unroll
      for (int j = 0; j < 16; ++j)
        b[j] = cbh[c * HD + k0 + b_kidx(g, j)];
      acc = __builtin_amdgcn_wmma_f32_16x16x32_f16(false, a, false, b, (short)0, acc,
                                                   false, false);
    }
#pragma unroll
    for (int r = 0; r < 8; ++r)
      dist[(r + 8 * g) * NCPAD + tn * 16 + lr] = acc[r];
  }
  __syncthreads();

  if (lane < 16) {                    // argmin_c (|cb_c|^2 - 2 f.cb_c)
    float best = 3.4e38f; int bi = 0;
    for (int c = 0; c < NCODE; ++c) {
      const float d = cbn2[c] - 2.0f * dist[lane * NCPAD + c];
      if (d < best) { best = d; bi = c; }
    }
    ridx[lane] = bi;
    idxf[row0 + lane] = (float)bi;
  }
  __syncthreads();

  // rotation trick, fp32, 32 lanes x 4 elems per 128-dim row
  float lsum = 0.f;
  for (int r = 0; r < 16; ++r) {
    const size_t grow = (size_t)(row0 + r);
    const int ci = ridx[r];
    float f[4], c4[4];
#pragma unroll
    for (int i = 0; i < 4; ++i) {
      f[i]  = xa[grow * HD + lane + 32 * i];
      c4[i] = cb[(size_t)ci * HD + lane + 32 * i];
    }
    float ff = 0.f, cc = 0.f;
#pragma unroll
    for (int i = 0; i < 4; ++i) { ff += f[i] * f[i]; cc += c4[i] * c4[i]; }
    ff = wave_sum(ff); cc = wave_sum(cc);
    const float nsrc = sqrtf(ff) + 1e-6f;
    const float ntgt = sqrtf(cc) + 1e-6f;

    float w[4], s1 = 0.f, s2 = 0.f;
#pragma unroll
    for (int i = 0; i < 4; ++i) {
      const float u  = f[i]  / nsrc;
      const float qh = c4[i] / ntgt;
      w[i] = u + qh;
      s1 += w[i] * w[i];
      s2 += f[i] * w[i];
    }
    s1 = wave_sum(s1); s2 = wave_sum(s2);
    const float en    = sqrtf(s1) + 1e-6f;     // ||u+qh|| + eps
    const float dfe   = s2 / en;               // f . e
    const float dfu   = ff / nsrc;             // f . u
    const float scale = ntgt / nsrc;
#pragma unroll
    for (int i = 0; i < 4; ++i) {
      const float e  = w[i] / en;
      const float qh = c4[i] / ntgt;
      q[grow * HD + lane + 32 * i] = (f[i] - 2.0f * dfe * e + 2.0f * dfu * qh) * scale;
      const float d = f[i] - c4[i];
      lsum += d * d;
    }
  }
  lsum = wave_sum(lsum);
  if (lane == 0) atomicAdd(loss, lsum * (1.0f / ((float)NROW * (float)HD)));
}

// ---------------------------------------------------------------------------
// launch
// ---------------------------------------------------------------------------
extern "C" void kernel_launch(void* const* d_in, const int* in_sizes, int n_in,
                              void* d_out, int out_size, void* d_ws, size_t ws_size,
                              hipStream_t stream) {
  const float* x  = (const float*)d_in[0];
  const float* gu = (const float*)d_in[1];
  const float* Wq = (const float*)d_in[2];
  const float* Wk = (const float*)d_in[3];
  const float* ce = (const float*)d_in[4];
  const float* Wc = (const float*)d_in[5];

  float* out  = (float*)d_out;
  float* qout = out;                       // [8192,2,128]
  float* idxf = out + (size_t)NROW * HD;   // [8192,2] as float
  float* loss = idxf + NROW;               // scalar

  char* ws = (char*)d_ws;
  _Float16* Mht  = (_Float16*)(ws + 0);        // 2*128*128 f16 =  65536 B
  float*    cb   = (float*)   (ws + 65536);    // 112*128 f32   =  57344 B
  _Float16* cbh  = (_Float16*)(ws + 122880);   // 112*128 f16   =  28672 B
  float*    cbn2 = (float*)   (ws + 151552);   // 112 f32
  float*    pet  = (float*)   (ws + 152064);   // 29*128 f32    =  14848 B
  float*    xa   = (float*)   (ws + 167936);   // 16384*128 f32 = 8 MB

  k_prep_M  <<<NHEAD * 64, 32, 0, stream>>>(Wq, Wk, Mht);
  k_prep_cb <<<56,         32, 0, stream>>>(ce, Wc, cb, cbh);
  k_prep_aux<<<1,         128, 0, stream>>>(cb, cbn2, pet, loss);
  k_attend  <<<NBAT / GB, 256, 0, stream>>>(x, gu, Mht, pet, xa);
  k_vq      <<<NROW / 16,  32, 0, stream>>>(xa, cb, cbh, cbn2, qout, idxf, loss);
}
